// SimpleRNN_16939351016020
// MI455X (gfx1250) — matrix operand order repaired and checked by reading, not compiled, not run on hardware
//
#include <hip/hip_runtime.h>

// Problem dims (from reference): BATCH=256, SEQ=512, IN=64, HID=512
#define B_  256
#define S_  512
#define IN_ 64
#define H_  512

typedef unsigned short u16;
typedef __attribute__((ext_vector_type(16))) u16    v16u;
typedef __attribute__((ext_vector_type(8)))  u16    v8u;
typedef __attribute__((ext_vector_type(16))) __bf16 v16bf;
typedef __attribute__((ext_vector_type(8)))  float  v8f;

union BFV { v16u u; v16bf b; };

// f32 -> bf16 round-to-nearest-even, pure bit manipulation (no __bf16 arith needed)
__device__ __forceinline__ u16 f2bf(float f) {
  union { float f; unsigned u; } v; v.f = f;
  unsigned r = (v.u + 0x7FFFu + ((v.u >> 16) & 1u)) >> 16;
  return (u16)r;
}

// ---------------------------------------------------------------------------
// Prep: convert W_hh, W_ih to bf16; seed bf16 h0 ping buffer.
// ---------------------------------------------------------------------------
__global__ __launch_bounds__(256)
void rnn_prep(const float* __restrict__ Whh, const float* __restrict__ Wih,
              const float* __restrict__ h0,
              u16* __restrict__ whh_bf, u16* __restrict__ wih_bf,
              u16* __restrict__ hbf0) {
  int i = blockIdx.x * blockDim.x + threadIdx.x;
  if (i < H_ * H_)  whh_bf[i] = f2bf(Whh[i]);
  if (i < H_ * IN_) wih_bf[i] = f2bf(Wih[i]);
  if (i < B_ * H_)  hbf0[i]   = f2bf(h0[i]);
}

// ---------------------------------------------------------------------------
// One recurrence step:  Hnew = tanh( x[:,t,:] @ W_ih^T + b + Hold @ W_hh^T )
// Grid: 64 blocks x 256 threads = 512 waves; one 16x16 output tile per wave.
// K-loop: 2 chunks of 32 over the input projection + 16 chunks over hidden.
// ---------------------------------------------------------------------------
__global__ __launch_bounds__(256)
void rnn_step(const float* __restrict__ x,
              const float* __restrict__ b_ih, const float* __restrict__ b_hh,
              const u16* __restrict__ Whh, const u16* __restrict__ Wih,
              const u16* __restrict__ Hold,
              float* __restrict__ Hf, u16* __restrict__ Hbf, int t) {
  const int wave = blockIdx.x * 8 + (threadIdx.x >> 5);
  const int lane = threadIdx.x & 31;
  const int tm   = wave & 15;        // 16 tiles over batch (256/16)
  const int tn   = wave >> 4;        // 32 tiles over hidden (512/16)
  const int l15  = lane & 15;
  const int hi   = lane >> 4;        // lane half select
  const int base0 = hi << 3;         // unused; A loads also use kb (0 or 16)
  const int kb    = hi << 4;         // B-fragment K sub-offset (0 or 16)
  const int m = tm * 16 + l15;       // batch row this lane loads for A
  const int n = tn * 16 + l15;       // hidden index: B row of W / output col

  v8f c = {0.f, 0.f, 0.f, 0.f, 0.f, 0.f, 0.f, 0.f};

  // ---- fused input projection: x[:, t, :] @ W_ih^T  (K = 64) ----
  const float* xrow = x + (size_t)m * (S_ * IN_) + (size_t)t * IN_;
#pragma unroll
  for (int kk = 0; kk < IN_; kk += 32) {
    v8f xlo = *(const v8f*)(xrow + kk + kb);           // K = kk+kb .. +7
    v8f xhi = *(const v8f*)(xrow + kk + kb + 8);       // K = kk+kb+8 .. +7
    BFV a;
#pragma unroll
    for (int i = 0; i < 8; ++i) { a.u[i] = f2bf(xlo[i]); a.u[8 + i] = f2bf(xhi[i]); }
    BFV b; b.u = *(const v16u*)(Wih + (size_t)n * IN_ + kk + kb);
    c = __builtin_amdgcn_wmma_f32_16x16x32_bf16(false, a.b, false, b.b,
                                                (short)0, c, false, false);
  }

  // ---- recurrence: Hold @ W_hh^T  (K = 512, bf16 inputs prestaged) ----
  const u16* hrow = Hold + (size_t)m * H_;
  const u16* wrow = Whh  + (size_t)n * H_;
#pragma unroll 4
  for (int kk = 0; kk < H_; kk += 32) {
    v8u lo = *(const v8u*)(hrow + kk + kb);
    v8u hh = *(const v8u*)(hrow + kk + kb + 8);
    BFV a; a.u = __builtin_shufflevector(lo, hh, 0, 1, 2, 3, 4, 5, 6, 7,
                                         8, 9, 10, 11, 12, 13, 14, 15);
    BFV b; b.u = *(const v16u*)(wrow + kk + kb);
    c = __builtin_amdgcn_wmma_f32_16x16x32_bf16(false, a.b, false, b.b,
                                                (short)0, c, false, false);
  }

  // ---- epilogue: bias + tanh; store f32 (output path) + bf16 (next step) ----
  const float bias = b_ih[n] + b_hh[n];
#pragma unroll
  for (int r = 0; r < 8; ++r) {
    const int row = tm * 16 + r + (hi << 3);           // C/D layout row
    const float v = tanhf(c[r] + bias);
    Hf [(size_t)row * H_ + n] = v;
    Hbf[(size_t)row * H_ + n] = f2bf(v);
  }
}

// ---------------------------------------------------------------------------
// Output = (h_n, h_n): duplicate the final f32 hidden state.
// ---------------------------------------------------------------------------
__global__ __launch_bounds__(256)
void rnn_finalize(const float* __restrict__ hf, float* __restrict__ out) {
  int i = blockIdx.x * blockDim.x + threadIdx.x;
  if (i < B_ * H_) {
    float v = hf[i];
    out[i] = v;
    out[B_ * H_ + i] = v;
  }
}

extern "C" void kernel_launch(void* const* d_in, const int* in_sizes, int n_in,
                              void* d_out, int out_size, void* d_ws, size_t ws_size,
                              hipStream_t stream) {
  const float* x   = (const float*)d_in[0];
  const float* h0  = (const float*)d_in[1];
  const float* Wih = (const float*)d_in[2];
  const float* Whh = (const float*)d_in[3];
  const float* bih = (const float*)d_in[4];
  const float* bhh = (const float*)d_in[5];
  float* out = (float*)d_out;

  // Workspace layout (all offsets 256-B aligned; ~1.6 MB total)
  char* ws = (char*)d_ws;
  u16*   whh_bf = (u16*)(ws + 0);         // 512*512*2 = 524288
  u16*   wih_bf = (u16*)(ws + 524288);    // 512*64*2  =  65536
  u16*   hbf0   = (u16*)(ws + 589824);    // 256*512*2 = 262144
  u16*   hbf1   = (u16*)(ws + 851968);    // 256*512*2 = 262144
  float* hf     = (float*)(ws + 1114112); // 256*512*4 = 524288

  rnn_prep<<<1024, 256, 0, stream>>>(Whh, Wih, h0, whh_bf, wih_bf, hbf0);

  for (int t = 0; t < S_; ++t) {
    const u16* hin = (t & 1) ? hbf1 : hbf0;
    u16*      hout = (t & 1) ? hbf0 : hbf1;
    rnn_step<<<64, 256, 0, stream>>>(x, bih, bhh, whh_bf, wih_bf, hin, hf, hout, t);
  }

  rnn_finalize<<<512, 256, 0, stream>>>(hf, out);
}
